// LSTM_22428319220608
// MI455X (gfx1250) — matrix operand order repaired and checked by reading, not compiled, not run on hardware
//
#include <hip/hip_runtime.h>

typedef __attribute__((ext_vector_type(16))) _Float16 v16h;
typedef __attribute__((ext_vector_type(8)))  _Float16 v8h;
typedef __attribute__((ext_vector_type(8)))  float    v8f;

#define BB 1024
#define TSEQ 336

// Fast activations: v_exp_f32 + v_rcp_f32 (denominator >= 1, so rcp is safe).
__device__ __forceinline__ float fsig(float x) {
  return __builtin_amdgcn_rcpf(1.0f + __expf(-x));
}
__device__ __forceinline__ float ftanh(float x) {
  x = fminf(fmaxf(x, -15.0f), 15.0f);
  float e = __expf(2.0f * x);
  return (e - 1.0f) * __builtin_amdgcn_rcpf(e + 1.0f);
}

// gfx1250 async copy: 16B global -> LDS per lane, tracked by ASYNCcnt.
// Generic LDS pointers keep the wave-relative LDS byte offset in bits [31:0].
__device__ __forceinline__ void async_copy16(unsigned lds_off, const void* g) {
  asm volatile("global_load_async_to_lds_b128 %0, %1, off"
               :: "v"(lds_off), "v"(g) : "memory");
}
__device__ __forceinline__ void async_wait0() {
  asm volatile("s_wait_asynccnt 0x0" ::: "memory");
}

// Build a 16x32 f16 A-fragment row for this lane.
// A k order used here (wave32): lane = 16*half + m; elements 0..7 <-> K=16*half+0..7,
// elements 8..15 <-> K=16*half+8..15 (relative to the 32-wide K block).
__device__ __forceinline__ v16h make_afrag(const _Float16* p0, const _Float16* p1) {
  v8h lo = *(const v8h*)p0;
  v8h hi = *(const v8h*)p1;
  v16h a;
#pragma unroll
  for (int i = 0; i < 8; ++i) { a[i] = lo[i]; a[i + 8] = hi[i]; }
  return a;
}

// x: f32 [B*T][16] -> xb: f16 [B*T][32], cols 16..31 zeroed (K padding for WMMA)
__global__ void cvt_x_kernel(const float* __restrict__ x, _Float16* __restrict__ xb, int nrows) {
  int i = blockIdx.x * blockDim.x + threadIdx.x;
  if (i >= nrows * 32) return;
  int row = i >> 5, k = i & 31;
  float v = (k < 16) ? x[row * 16 + k] : 0.0f;
  xb[i] = (_Float16)v;
}

template<int DIN, int DINP, int H, int KB_IN, int KB_H, int NCH, int OUTS>
struct ScanCfg {
  static constexpr int G   = 4 * H;       // gate dim
  static constexpr int GP  = NCH * 16;    // padded gate dim (one 16-wide N chunk per wave)
  static constexpr int HPK = KB_H * 32;   // padded hidden dim (K blocks of 32)
  static constexpr int NTH = NCH * 32;    // one wave per N chunk
  static constexpr size_t b_whh  = (size_t)GP * HPK * 2;   // f16 [GP][HPK]
  static constexpr size_t b_wih  = (size_t)GP * DINP * 2;  // f16 [GP][DINP]
  static constexpr size_t b_h    = (size_t)16 * HPK * 2;   // f16 [16][HPK]
  static constexpr size_t b_x1   = (size_t)16 * DINP * 2;  // one x stage buffer
  static constexpr size_t b_bias = (size_t)GP * 4;         // f32 [GP]
  static constexpr size_t b_g    = (size_t)16 * GP * 4;    // f32 [16][GP]
  static constexpr size_t b_c    = (size_t)16 * H * 4;     // f32 [16*H]
  static constexpr size_t lds = b_whh + b_wih + b_h + 2 * b_x1 + b_bias + b_g + b_c;
};

// Persistent single-direction LSTM scan. One WG = 16 batch rows, all T steps.
// grid = (B/16, 2): blockIdx.y selects direction (0=fw, 1=bw/reversed time).
// xin : f16 [B][T][DINP] (zero-padded cols DIN..DINP-1)
// hout: f16 [B][T][OUTS]; this dir writes cols [dir*H, dir*H+H); dir 0 zeroes pad cols [2H, OUTS)
template<int DIN, int DINP, int H, int KB_IN, int KB_H, int NCH, int OUTS>
__global__ __launch_bounds__(NCH * 32)
void lstm_scan(const _Float16* __restrict__ xin, _Float16* __restrict__ hout,
               const float* __restrict__ fwih, const float* __restrict__ fwhh,
               const float* __restrict__ fbih, const float* __restrict__ fbhh,
               const float* __restrict__ bwih, const float* __restrict__ bwhh,
               const float* __restrict__ bbih, const float* __restrict__ bbhh) {
  using C = ScanCfg<DIN, DINP, H, KB_IN, KB_H, NCH, OUTS>;
  constexpr int G = C::G, GP = C::GP, HPK = C::HPK, NTH = C::NTH;
  constexpr int EW   = 16 * H;                     // element-wise work items
  constexpr int JMAX = (EW + NTH - 1) / NTH;
  constexpr int PW   = OUTS - 2 * H;               // pad cols
  constexpr int PMAX = (16 * PW + NTH - 1) / NTH;
  constexpr int XC   = (16 * DINP) / 8;            // 16B x-stage chunks per timestep
  constexpr int XMAX = (XC + NTH - 1) / NTH;
  constexpr unsigned XBUFB = (unsigned)C::b_x1;    // byte distance between the 2 stage bufs

  extern __shared__ __align__(32) char smem[];
  _Float16* s_whh  = (_Float16*)(smem);
  _Float16* s_wih  = (_Float16*)(smem + C::b_whh);
  _Float16* s_h    = (_Float16*)(smem + C::b_whh + C::b_wih);
  _Float16* s_x0   = (_Float16*)(smem + C::b_whh + C::b_wih + C::b_h);
  float*    s_bias = (float*)(smem + C::b_whh + C::b_wih + C::b_h + 2 * C::b_x1);
  float*    s_g    = (float*)(smem + C::b_whh + C::b_wih + C::b_h + 2 * C::b_x1 + C::b_bias);
  float*    s_c    = (float*)(smem + C::b_whh + C::b_wih + C::b_h + 2 * C::b_x1 + C::b_bias + C::b_g);

  const int tid = threadIdx.x;
  const int dir = blockIdx.y;
  const int b0  = blockIdx.x * 16;
  const int t0  = dir ? (TSEQ - 1) : 0;
  const int ostep = dir ? -OUTS : OUTS;                       // hout elems per t step
  const long xstep = dir ? -(long)DINP : (long)DINP;          // xin elems per t step

  // ---- per-thread async x-stage chunks (16B each), initially at t0 ----
  const _Float16* xg[XMAX];
  unsigned xl0[XMAX];
  bool xv[XMAX];
#pragma unroll
  for (int q = 0; q < XMAX; ++q) {
    int cidx = tid + q * NTH;
    xv[q] = cidx < XC;
    int cc = xv[q] ? cidx : 0;
    int m = cc / (DINP / 8), o = (cc % (DINP / 8)) * 8;
    xg[q]  = xin + ((size_t)(b0 + m) * TSEQ + t0) * DINP + o;
    xl0[q] = (unsigned)(uintptr_t)&s_x0[m * DINP + o];
  }
  // kick off the t0 stage into buffer 0; it overlaps the weight staging below
#pragma unroll
  for (int q = 0; q < XMAX; ++q)
    if (xv[q]) async_copy16(xl0[q], xg[q]);
#pragma unroll
  for (int q = 0; q < XMAX; ++q) xg[q] += xstep;  // now points at t1

  const float* wih = dir ? bwih : fwih;
  const float* whh = dir ? bwhh : fwhh;
  const float* bih = dir ? bbih : fbih;
  const float* bhh = dir ? bbhh : fbhh;

  // ---- stage weights into LDS as zero-padded f16 (one-time; reused 336x) ----
  for (int i = tid; i < GP * DINP; i += NTH) {
    int g = i / DINP, k = i % DINP;
    float v = (g < G && k < DIN) ? wih[g * DIN + k] : 0.0f;
    s_wih[i] = (_Float16)v;
  }
  for (int i = tid; i < GP * HPK; i += NTH) {
    int g = i / HPK, k = i % HPK;
    float v = (g < G && k < H) ? whh[g * H + k] : 0.0f;
    s_whh[i] = (_Float16)v;
  }
  for (int i = tid; i < GP; i += NTH) s_bias[i] = (i < G) ? (bih[i] + bhh[i]) : 0.0f;
  for (int i = tid; i < 16 * HPK; i += NTH) s_h[i] = (_Float16)0.0f;  // h0 = 0 (+K pads)
  for (int i = tid; i < 16 * H;  i += NTH) s_c[i] = 0.0f;             // c0 = 0
  async_wait0();
  __syncthreads();

  // ---- preload this wave's B-fragments (weights) into registers ----
  // B layout (K x 16): lane = 16*kh + n', elements 0..15 <-> K = 16*kh + 0..15 per 32-K block
  const int lane = tid & 31;
  const int wave = tid >> 5;
  const int nl = lane & 15;       // N within chunk == A-row m this lane loads
  const int kh = lane >> 4;       // K-half selector
  const int n  = wave * 16 + nl;  // global gate column

  v16h wihf[KB_IN];
#pragma unroll
  for (int kb = 0; kb < KB_IN; ++kb)
    wihf[kb] = *(const v16h*)&s_wih[n * DINP + kb * 32 + kh * 16];
  v16h whhf[KB_H];
#pragma unroll
  for (int kb = 0; kb < KB_H; ++kb)
    whhf[kb] = *(const v16h*)&s_whh[n * HPK + kb * 32 + kh * 16];
  const float bias_n = s_bias[n];

  // ---- per-thread element-wise bookkeeping; output ptrs stepped by +-OUTS ----
  int ewj[JMAX], ewm[JMAX], ewjj[JMAX];
  bool ewv[JMAX];
  _Float16* ewp[JMAX];
#pragma unroll
  for (int q = 0; q < JMAX; ++q) {
    int j = tid + q * NTH;
    ewv[q] = j < EW;
    int jc = ewv[q] ? j : 0;
    ewm[q] = jc & 15; ewjj[q] = jc >> 4; ewj[q] = jc;
    ewp[q] = hout + ((size_t)(b0 + ewm[q]) * TSEQ + t0) * OUTS + (size_t)dir * H + ewjj[q];
  }
  bool pv[PMAX];
  _Float16* ppt[PMAX];
#pragma unroll
  for (int q = 0; q < PMAX; ++q) {
    int j = tid + q * NTH;
    pv[q] = (dir == 0) && (j < 16 * PW);
    int jc = (j < 16 * PW) ? j : 0;
    int m = jc & 15, pc = 2 * H + (jc >> 4);
    ppt[q] = hout + ((size_t)(b0 + m) * TSEQ + t0) * OUTS + pc;
  }

  // read-buffer byte offset inside the double buffer; toggled by XOR each step
  unsigned roff = 0;

  // ---- sequential scan over time ----
  for (int tt = 0; tt < TSEQ; ++tt) {
    const _Float16* xr = (const _Float16*)((const char*)s_x0 + roff) + nl * DINP;
    const _Float16* hr = s_h + nl * HPK;

    // load A-fragments; loads for fragment k+1 overlap WMMA k on the XDL pipe
    v16h ax[KB_IN];
#pragma unroll
    for (int kb = 0; kb < KB_IN; ++kb)
      ax[kb] = make_afrag(xr + kb * 32 + kh * 16, xr + kb * 32 + kh * 16 + 8);
    v16h ah[KB_H];
#pragma unroll
    for (int kb = 0; kb < KB_H; ++kb)
      ah[kb] = make_afrag(hr + kb * 32 + kh * 16, hr + kb * 32 + kh * 16 + 8);

    v8f acc;
#pragma unroll
    for (int r = 0; r < 8; ++r) acc[r] = bias_n;  // bias pre-loaded into C
#pragma unroll
    for (int kb = 0; kb < KB_IN; ++kb)
      acc = __builtin_amdgcn_wmma_f32_16x16x32_f16(false, ax[kb], false, wihf[kb],
                                                   (short)0, acc, false, false);
#pragma unroll
    for (int kb = 0; kb < KB_H; ++kb)
      acc = __builtin_amdgcn_wmma_f32_16x16x32_f16(false, ah[kb], false, whhf[kb],
                                                   (short)0, acc, false, false);

    // spill gate pre-activations: C layout VGPR r -> M = r + 8*kh, N = n
#pragma unroll
    for (int r = 0; r < 8; ++r) s_g[(r + 8 * kh) * GP + n] = acc[r];

    // software pipeline: async-stage x_{t+1} into the buffer we are NOT reading
    if (tt + 1 < TSEQ) {
      const unsigned woff = roff ^ XBUFB;
#pragma unroll
      for (int q = 0; q < XMAX; ++q)
        if (xv[q]) async_copy16(xl0[q] + woff, xg[q]);
    }
#pragma unroll
    for (int q = 0; q < XMAX; ++q) xg[q] += xstep;

    __syncthreads();

    // ---- element-wise LSTM cell update (f32) ----
#pragma unroll
    for (int q = 0; q < JMAX; ++q) {
      if (ewv[q]) {
        const int m = ewm[q], jj = ewjj[q];
        float gi = s_g[m * GP + jj];
        float gf = s_g[m * GP + H + jj];
        float gg = s_g[m * GP + 2 * H + jj];
        float go = s_g[m * GP + 3 * H + jj];
        float iv = fsig(gi), fv = fsig(gf), gv = ftanh(gg), ov = fsig(go);
        float cn = fv * s_c[ewj[q]] + iv * gv;
        s_c[ewj[q]] = cn;
        float hn = ov * ftanh(cn);
        _Float16 h16 = (_Float16)hn;
        s_h[m * HPK + jj] = h16;
        *ewp[q] = h16;
      }
      ewp[q] += ostep;
    }
#pragma unroll
    for (int q = 0; q < PMAX; ++q) {   // zero K-pad cols of this (b,t) row
      if (pv[q]) *ppt[q] = (_Float16)0.0f;
      ppt[q] += ostep;
    }

    async_wait0();      // our x_{t+1} chunks landed in LDS
    roff ^= XBUFB;      // swap read/write stage buffers
    __syncthreads();    // publish h, c, gates, x stage to whole WG
  }
}

// MLP head: relu(h@w1^T+b1) -> relu(@w2^T+b2) -> @wo^T+bo. One thread per (b,t).
__global__ void mlp_head(const _Float16* __restrict__ h2,
                         const float* __restrict__ w1, const float* __restrict__ b1,
                         const float* __restrict__ w2, const float* __restrict__ b2,
                         const float* __restrict__ wo, const float* __restrict__ bo,
                         float* __restrict__ out, int n) {
  __shared__ float s_w1[30 * 42];
  __shared__ float s_b1[30];
  __shared__ float s_w2[20 * 30];
  __shared__ float s_b2[20];
  __shared__ float s_wo[20];
  __shared__ float s_bo;
  const int tid = threadIdx.x;
  for (int i = tid; i < 30 * 42; i += blockDim.x) s_w1[i] = w1[i];
  for (int i = tid; i < 20 * 30; i += blockDim.x) s_w2[i] = w2[i];
  for (int i = tid; i < 30; i += blockDim.x) s_b1[i] = b1[i];
  for (int i = tid; i < 20; i += blockDim.x) { s_b2[i] = b2[i]; s_wo[i] = wo[i]; }
  if (tid == 0) s_bo = bo[0];
  __syncthreads();

  int i = blockIdx.x * blockDim.x + tid;
  if (i >= n) return;
  const _Float16* hp = h2 + (size_t)i * 48;
  union { v8h v; _Float16 e[8]; } u[6];
#pragma unroll
  for (int q = 0; q < 6; ++q) u[q].v = *(const v8h*)(hp + q * 8);
  float hv[42];
#pragma unroll
  for (int k = 0; k < 42; ++k) hv[k] = (float)u[k >> 3].e[k & 7];

  float a2[20];
#pragma unroll
  for (int o = 0; o < 20; ++o) a2[o] = s_b2[o];
  for (int o1 = 0; o1 < 30; ++o1) {
    float s = s_b1[o1];
#pragma unroll
    for (int k = 0; k < 42; ++k) s = fmaf(s_w1[o1 * 42 + k], hv[k], s);
    s = fmaxf(s, 0.0f);
#pragma unroll
    for (int o = 0; o < 20; ++o) a2[o] = fmaf(s_w2[o * 30 + o1], s, a2[o]);
  }
  float r = s_bo;
#pragma unroll
  for (int o = 0; o < 20; ++o) r = fmaf(s_wo[o], fmaxf(a2[o], 0.0f), r);
  out[i] = r;
}

extern "C" void kernel_launch(void* const* d_in, const int* in_sizes, int n_in,
                              void* d_out, int out_size, void* d_ws, size_t ws_size,
                              hipStream_t stream) {
  (void)in_sizes; (void)n_in; (void)out_size; (void)ws_size;
  const float* x = (const float*)d_in[0];

  // workspace layout (f16): xb [B*T][32] | h1 [B*T][160] | h2 [B*T][48]
  char* ws = (char*)d_ws;
  const size_t xb_b = (size_t)BB * TSEQ * 32 * sizeof(_Float16);   // ~22 MB
  const size_t h1_b = (size_t)BB * TSEQ * 160 * sizeof(_Float16);  // ~110 MB
  _Float16* xb = (_Float16*)ws;
  _Float16* h1 = (_Float16*)(ws + xb_b);
  _Float16* h2 = (_Float16*)(ws + xb_b + h1_b);

  const int nrows = BB * TSEQ;

  using C1 = ScanCfg<16, 32, 70, 1, 3, 18, 160>;   // L1: F=16, H=70, 18 gate chunks
  using C2 = ScanCfg<140, 160, 21, 5, 1, 6, 48>;   // L2: DIN=140, H=21, 6 gate chunks

  // dynamic LDS can exceed the 64KB default cap -> opt in (host-side, capture-safe)
  hipFuncSetAttribute((const void*)&lstm_scan<16, 32, 70, 1, 3, 18, 160>,
                      hipFuncAttributeMaxDynamicSharedMemorySize, (int)C1::lds);
  hipFuncSetAttribute((const void*)&lstm_scan<140, 160, 21, 5, 1, 6, 48>,
                      hipFuncAttributeMaxDynamicSharedMemorySize, (int)C2::lds);

  cvt_x_kernel<<<(nrows * 32 + 255) / 256, 256, 0, stream>>>(x, xb, nrows);

  lstm_scan<16, 32, 70, 1, 3, 18, 160>
      <<<dim3(BB / 16, 2), C1::NTH, C1::lds, stream>>>(
          xb, h1,
          (const float*)d_in[1], (const float*)d_in[2],
          (const float*)d_in[3], (const float*)d_in[4],
          (const float*)d_in[5], (const float*)d_in[6],
          (const float*)d_in[7], (const float*)d_in[8]);

  lstm_scan<140, 160, 21, 5, 1, 6, 48>
      <<<dim3(BB / 16, 2), C2::NTH, C2::lds, stream>>>(
          h1, h2,
          (const float*)d_in[9],  (const float*)d_in[10],
          (const float*)d_in[11], (const float*)d_in[12],
          (const float*)d_in[13], (const float*)d_in[14],
          (const float*)d_in[15], (const float*)d_in[16]);

  mlp_head<<<(nrows + 255) / 256, 256, 0, stream>>>(
      h2,
      (const float*)d_in[17], (const float*)d_in[18],
      (const float*)d_in[19], (const float*)d_in[20],
      (const float*)d_in[21], (const float*)d_in[22],
      (float*)d_out, nrows);
}
